// GATorderedDeep_88175678587735
// MI455X (gfx1250) — compile-verified, matrix operand order checked by reading, not run on hardware
//
#include <hip/hip_runtime.h>
#include <hip/hip_bf16.h>

typedef __attribute__((ext_vector_type(16))) _Float16 v16h;
typedef __attribute__((ext_vector_type(8)))  _Float16 v8h;
typedef __attribute__((ext_vector_type(8)))  float    v8f;
typedef __attribute__((ext_vector_type(4)))  float    v4f;

#define MAXDEG 96

// ---------------------------------------------------------------------------
// Pack B (K x N, f32, row-major, head-batched) into f16 column-major
// Bt[((h*Npad)+n)*K + k], zero-padded for n >= N. Npad is a multiple of 32.
// ---------------------------------------------------------------------------
__global__ void k_pack_bT(const float* __restrict__ B, int ldb, long long bBatch,
                          _Float16* __restrict__ Bt, int K, int N, int Npad,
                          int heads)
{
  const long long gid = (long long)blockIdx.x * blockDim.x + threadIdx.x;
  const long long total = (long long)heads * Npad * K;
  if (gid >= total) return;
  const int k = (int)(gid % K);
  const long long t = gid / K;
  const int n = (int)(t % Npad);
  const int h = (int)(t / Npad);
  const float v = (n < N) ? B[(long long)h * bBatch + (long long)k * ldb + n] : 0.0f;
  Bt[gid] = (_Float16)v;
}

// ---------------------------------------------------------------------------
// WMMA GEMM: C[M x N] = gatherScale(A1)[M x K1] * Bt1^T  (+ A2 * Bt2^T) (+bias)
// A: f32 row-major (K1,K2 multiples of 32; M multiple of 16).
// Bt: f16 column-major (padded to Npad, multiple of 32).
// One wave per 16x32 output tile; heads batched over gridDim.z.
// Fragment layout per ISA: lane half selects kb in {0,8}; 16 f16 elements are
// two contiguous runs of 8 K-values -> pure b128 loads, no divergence.
// ---------------------------------------------------------------------------
__global__ void k_gemm_wmma(
    const float* __restrict__ A1, int lda1, long long a1Batch,
    const _Float16* __restrict__ Bt1, int K1, long long bt1Batch,
    const int* __restrict__ gidx, const float* __restrict__ gval,
    long long gBatch, int gStride,
    const float* __restrict__ A2, int lda2, long long a2Batch,
    const _Float16* __restrict__ Bt2, int K2, long long bt2Batch,
    const float* __restrict__ bias, long long biasBatch,
    float* __restrict__ C, int ldc, long long cBatch,
    int M, int N)
{
  const int bz = blockIdx.z;
  const int lane = threadIdx.x & 31;
  const int half = lane >> 4;      // 0: lanes 0-15, 1: lanes 16-31
  const int l15  = lane & 15;
  const int kb   = half * 8;       // K base of this lane's fragment runs
  const int tm = blockIdx.y * 16;
  const int tn = blockIdx.x * 32;

  // --- A1 row (optionally gathered + scaled) ---
  int r = tm + l15;
  float sc = 1.0f;
  if (gidx) {
    gidx += (long long)bz * gBatch;
    gval += (long long)bz * gBatch;
    sc = gval[(long long)r * gStride];
    r  = gidx[(long long)r * gStride];
  }
  const float* ap1 = A1 + (long long)bz * a1Batch + (long long)r * lda1;

  // --- Bt rows for the two 16-wide sub-tiles ---
  const _Float16* btb1 = Bt1 + (long long)bz * bt1Batch;
  const _Float16* b1c0 = btb1 + (long long)(tn + l15) * K1;
  const _Float16* b1c1 = btb1 + (long long)(tn + 16 + l15) * K1;

  v8f acc0 = {};
  v8f acc1 = {};

  auto mm = [&](const float* ap, float s, const _Float16* bc0,
                const _Float16* bc1, int K) {
    for (int k0 = 0; k0 < K; k0 += 32) {
      const v4f x0 = *(const v4f*)(ap + k0 + kb);
      const v4f x1 = *(const v4f*)(ap + k0 + kb + 4);
      const v4f x2 = *(const v4f*)(ap + k0 + kb + 16);
      const v4f x3 = *(const v4f*)(ap + k0 + kb + 20);
      v16h af;
#pragma unroll
      for (int i = 0; i < 4; ++i) {
        af[i]      = (_Float16)(x0[i] * s);
        af[4 + i]  = (_Float16)(x1[i] * s);
        af[8 + i]  = (_Float16)(x2[i] * s);
        af[12 + i] = (_Float16)(x3[i] * s);
      }
      const v8h l0 = *(const v8h*)(bc0 + k0 + kb);
      const v8h h0 = *(const v8h*)(bc0 + k0 + kb + 16);
      const v16h bf0 = __builtin_shufflevector(
          l0, h0, 0, 1, 2, 3, 4, 5, 6, 7, 8, 9, 10, 11, 12, 13, 14, 15);
      acc0 = __builtin_amdgcn_wmma_f32_16x16x32_f16(
          false, af, false, bf0, (short)0, acc0, false, false);
      const v8h l1 = *(const v8h*)(bc1 + k0 + kb);
      const v8h h1 = *(const v8h*)(bc1 + k0 + kb + 16);
      const v16h bf1 = __builtin_shufflevector(
          l1, h1, 0, 1, 2, 3, 4, 5, 6, 7, 8, 9, 10, 11, 12, 13, 14, 15);
      acc1 = __builtin_amdgcn_wmma_f32_16x16x32_f16(
          false, af, false, bf1, (short)0, acc1, false, false);
    }
  };

  mm(ap1, sc, b1c0, b1c1, K1);

  if (A2) {
    const float* ap2 = A2 + (long long)bz * a2Batch + (long long)(tm + l15) * lda2;
    const _Float16* btb2 = Bt2 + (long long)bz * bt2Batch;
    mm(ap2, 1.0f, btb2 + (long long)(tn + l15) * K2,
       btb2 + (long long)(tn + 16 + l15) * K2, K2);
  }

  float* Cb = C + (long long)bz * cBatch;
  const float* bb = bias ? bias + (long long)bz * biasBatch : nullptr;
#pragma unroll
  for (int v = 0; v < 8; ++v) {
    const int m = tm + v + half * 8;
    const int n0 = tn + l15;
    const int n1 = n0 + 16;
    if (n0 < N) Cb[(long long)m * ldc + n0] = acc0[v] + (bb ? bb[n0] : 0.0f);
    if (n1 < N) Cb[(long long)m * ldc + n1] = acc1[v] + (bb ? bb[n1] : 0.0f);
  }
}

// ---------------------------------------------------------------------------
// Compact adjacency rows into neighbor index lists (one wave per row).
// ---------------------------------------------------------------------------
__global__ void k_neighbors(const float* __restrict__ adj, int N,
                            int* __restrict__ nidx, int* __restrict__ ncnt)
{
  __shared__ int cnt;
  const int row = blockIdx.x;
  const int lane = threadIdx.x;
  if (lane == 0) cnt = 0;
  __syncthreads();
  for (int j = lane; j < N; j += 32) {
    if (adj[(long long)row * N + j] > 0.0f) {
      int p = atomicAdd(&cnt, 1);
      if (p < MAXDEG) nidx[(long long)row * MAXDEG + p] = j;
    }
  }
  __syncthreads();
  if (lane == 0) ncnt[row] = cnt < MAXDEG ? cnt : MAXDEG;
}

// ---------------------------------------------------------------------------
// Per-(head,node) attention projections e1 = Wh.a[:d], e2 = Wh.a[d:]
// ---------------------------------------------------------------------------
__global__ void k_e1e2(const float* __restrict__ Wh, const float* __restrict__ a,
                       float* __restrict__ e1, float* __restrict__ e2,
                       int d, int N, int heads)
{
  const int gid = blockIdx.x * blockDim.x + threadIdx.x;
  if (gid >= heads * N) return;
  const int h = gid / N, node = gid - h * N;
  const float* wr = Wh + ((long long)h * N + node) * d;
  const float* ah = a + (long long)h * 2 * d;
  float s1 = 0.f, s2 = 0.f;
  for (int k = 0; k < d; ++k) { s1 += wr[k] * ah[k]; s2 += wr[k] * ah[d + k]; }
  e1[gid] = s1;
  e2[gid] = s2;
}

// ---------------------------------------------------------------------------
// Masked softmax over neighbors + top-16 (one wave per head,row). softmax is
// monotone so top-K of att == top-K of e; padded slots get val=0 (matches
// exp(NEG)=0 contribution in the reference).
// ---------------------------------------------------------------------------
__global__ void k_attn_topk(const float* __restrict__ e1, const float* __restrict__ e2,
                            const int* __restrict__ nidx, const int* __restrict__ ncnt,
                            int* __restrict__ tidx, float* __restrict__ tval, int N)
{
  const int row = blockIdx.x, head = blockIdx.y;
  const int lane = threadIdx.x;
  __shared__ float sval[MAXDEG];
  __shared__ int   sidx[MAXDEG];
  const int deg = ncnt[row];
  const float my_e1 = e1[(long long)head * N + row];

  float mx = -1e30f;
  for (int p = lane; p < deg; p += 32) {
    const int j = nidx[(long long)row * MAXDEG + p];
    float e = my_e1 + e2[(long long)head * N + j];
    e = (e > 0.f) ? e : 0.2f * e;               // LeakyReLU(0.2)
    sval[p] = e; sidx[p] = j;
    mx = fmaxf(mx, e);
  }
  for (int o = 16; o > 0; o >>= 1) mx = fmaxf(mx, __shfl_xor(mx, o, 32));
  __syncthreads();
  float sum = 0.f;
  for (int p = lane; p < deg; p += 32) {
    const float wgt = __expf(sval[p] - mx);
    sval[p] = wgt; sum += wgt;
  }
  for (int o = 16; o > 0; o >>= 1) sum += __shfl_xor(sum, o, 32);
  const float inv = 1.0f / sum;                 // deg >= 1 (self loop)
  __syncthreads();
  if (lane == 0) {
    for (int s = 0; s < 16; ++s) {
      float best = -1.0f; int bp = -1;
      for (int p = 0; p < deg; ++p)
        if (sval[p] > best) { best = sval[p]; bp = p; }
      const long long o = ((long long)head * N + row) * 16 + s;
      if (bp >= 0) { tidx[o] = sidx[bp]; tval[o] = best * inv; sval[bp] = -2.0f; }
      else         { tidx[o] = 0;        tval[o] = 0.0f; }
    }
  }
}

// ---------------------------------------------------------------------------
// LSTM pointwise gates: z=(i,f,g,o) blocks of 64; c plain per-head layout,
// h written in concat layout (node*hRow + head*hOff + j).
// ---------------------------------------------------------------------------
__global__ void k_lstm_gates(const float* __restrict__ z, float* __restrict__ c,
                             float* __restrict__ h, int N, int heads,
                             int hRow, int hOff)
{
  const int gid = blockIdx.x * blockDim.x + threadIdx.x;
  const int total = heads * N * 64;
  if (gid >= total) return;
  const int j = gid & 63;
  const int node = (gid >> 6) % N;
  const int head = gid / (64 * N);
  const float* zr = z + ((long long)head * N + node) * 256;
  const float zi = zr[j], zf = zr[64 + j], zg = zr[128 + j], zo = zr[192 + j];
  const long long ci = ((long long)head * N + node) * 64 + j;
  float cc = c[ci];
  const float si = 1.f / (1.f + __expf(-zi));
  const float sf = 1.f / (1.f + __expf(-zf));
  const float so = 1.f / (1.f + __expf(-zo));
  cc = sf * cc + si * tanhf(zg);
  c[ci] = cc;
  h[(long long)node * hRow + head * hOff + j] = so * tanhf(cc);
}

__global__ void k_elu(float* __restrict__ x, int n)
{
  const int i = blockIdx.x * blockDim.x + threadIdx.x;
  if (i < n) { const float v = x[i]; x[i] = (v > 0.f) ? v : (__expf(v) - 1.f); }
}

// ---------------------------------------------------------------------------
// Output head: att(row) over neighbors of Whp, elu, log_softmax. One wave/row.
// ---------------------------------------------------------------------------
__global__ void k_final(const float* __restrict__ Whp,
                        const float* __restrict__ eo1, const float* __restrict__ eo2,
                        const int* __restrict__ nidx, const int* __restrict__ ncnt,
                        float* __restrict__ out, int N, int C)
{
  const int row = blockIdx.x;
  const int lane = threadIdx.x;
  __shared__ float att[MAXDEG];
  __shared__ float acc[64];
  const int deg = ncnt[row];
  const float my = eo1[row];

  float mx = -1e30f;
  for (int p = lane; p < deg; p += 32) {
    const int j = nidx[(long long)row * MAXDEG + p];
    float e = my + eo2[j];
    e = (e > 0.f) ? e : 0.2f * e;
    att[p] = e; mx = fmaxf(mx, e);
  }
  for (int o = 16; o > 0; o >>= 1) mx = fmaxf(mx, __shfl_xor(mx, o, 32));
  __syncthreads();
  float sum = 0.f;
  for (int p = lane; p < deg; p += 32) {
    const float wgt = __expf(att[p] - mx);
    att[p] = wgt; sum += wgt;
  }
  for (int o = 16; o > 0; o >>= 1) sum += __shfl_xor(sum, o, 32);
  const float inv = 1.0f / sum;
  __syncthreads();

  for (int c = lane; c < C; c += 32) {
    float s = 0.f;
    for (int p = 0; p < deg; ++p) {
      const int j = nidx[(long long)row * MAXDEG + p];
      s += att[p] * inv * Whp[(long long)j * C + c];
    }
    acc[c] = (s > 0.f) ? s : (__expf(s) - 1.f);   // elu
  }
  __syncthreads();
  if (lane == 0) {
    float m = -1e30f;
    for (int c = 0; c < C; ++c) m = fmaxf(m, acc[c]);
    float se = 0.f;
    for (int c = 0; c < C; ++c) se += __expf(acc[c] - m);
    const float ls = logf(se);
    for (int c = 0; c < C; ++c) out[(long long)row * C + c] = acc[c] - m - ls;
  }
}

// ---------------------------------------------------------------------------
extern "C" void kernel_launch(void* const* d_in, const int* in_sizes, int n_in,
                              void* d_out, int out_size, void* d_ws, size_t ws_size,
                              hipStream_t stream)
{
  const float* x    = (const float*)d_in[0];
  const float* adj  = (const float*)d_in[1];
  const float* W1   = (const float*)d_in[2];
  const float* a1   = (const float*)d_in[3];
  const float* Wi1  = (const float*)d_in[4];
  const float* Wh1w = (const float*)d_in[5];
  const float* b1   = (const float*)d_in[6];
  const float* W2   = (const float*)d_in[7];
  const float* a2   = (const float*)d_in[8];
  const float* Wi2  = (const float*)d_in[9];
  const float* Wh2w = (const float*)d_in[10];
  const float* b2   = (const float*)d_in[11];
  const float* Wo   = (const float*)d_in[12];
  const float* ao   = (const float*)d_in[13];
  float* out = (float*)d_out;

  const int N = 2048, FIN = 1024, D = 64, H1 = 8, H2 = 4, NC = 40, KN = 16;

  char* wbase = (char*)d_ws;
  size_t off = 0;
  auto alloc = [&](size_t bytes) -> void* {
    off = (off + 255) & ~(size_t)255;
    void* p = wbase + off;
    off += bytes;
    return p;
  };

  int*   nidx  = (int*)  alloc((size_t)N * MAXDEG * 4);
  int*   ncnt  = (int*)  alloc((size_t)N * 4);
  float* Whbuf = (float*)alloc((size_t)H1 * N * D * 4);   // per-head Wh (reused L1/L2)
  float* e1b   = (float*)alloc((size_t)H1 * N * 4);
  float* e2b   = (float*)alloc((size_t)H1 * N * 4);
  int*   tidx  = (int*)  alloc((size_t)H1 * N * KN * 4);
  float* tval  = (float*)alloc((size_t)H1 * N * KN * 4);
  float* cbuf  = (float*)alloc((size_t)H1 * N * D * 4);
  float* h1cat = (float*)alloc((size_t)N * H1 * D * 4);   // N x 512
  float* h2cat = (float*)alloc((size_t)N * H2 * D * 4);   // N x 256
  float* zbuf  = (float*)alloc((size_t)H1 * N * 256 * 4);
  float* Whp   = (float*)alloc((size_t)N * NC * 4);

  // packed f16 column-major weights (Npad multiple of 32)
  _Float16* W1p  = (_Float16*)alloc((size_t)H1 * 64 * FIN * 2);       // 8 x 64 x 1024
  _Float16* Wi1p = (_Float16*)alloc((size_t)H1 * 256 * D * 2);        // 8 x 256 x 64
  _Float16* Wh1p = (_Float16*)alloc((size_t)H1 * 256 * D * 2);
  _Float16* W2p  = (_Float16*)alloc((size_t)H2 * 64 * (H1 * D) * 2);  // 4 x 64 x 512
  _Float16* Wi2p = (_Float16*)alloc((size_t)H2 * 256 * D * 2);
  _Float16* Wh2p = (_Float16*)alloc((size_t)H2 * 256 * D * 2);
  _Float16* Wop  = (_Float16*)alloc((size_t)64 * (H2 * D) * 2);       // 64 x 256

  auto packN = [&](const float* B, int ldb, long long bB, _Float16* Bt,
                   int K, int Nn, int Npad, int heads) {
    long long total = (long long)heads * Npad * K;
    k_pack_bT<<<(unsigned)((total + 255) / 256), 256, 0, stream>>>(
        B, ldb, bB, Bt, K, Nn, Npad, heads);
  };
  packN(W1,   D,   (long long)FIN * D,      W1p,  FIN,      D,   64,  H1);
  packN(Wi1,  256, (long long)D * 256,      Wi1p, D,        256, 256, H1);
  packN(Wh1w, 256, (long long)D * 256,      Wh1p, D,        256, 256, H1);
  packN(W2,   D,   (long long)(H1 * D) * D, W2p,  H1 * D,   D,   64,  H2);
  packN(Wi2,  256, (long long)D * 256,      Wi2p, D,        256, 256, H2);
  packN(Wh2w, 256, (long long)D * 256,      Wh2p, D,        256, 256, H2);
  packN(Wo,   NC,  0,                       Wop,  H2 * D,   NC,  64,  1);

  // ---- neighbor compaction (adj read exactly once) ----
  k_neighbors<<<N, 32, 0, stream>>>(adj, N, nidx, ncnt);

  // ================= Layer 1 (8 heads) =================
  k_gemm_wmma<<<dim3(64 / 32, N / 16, H1), 32, 0, stream>>>(
      x, FIN, 0, W1p, FIN, (long long)64 * FIN,
      nullptr, nullptr, 0, 0,
      nullptr, 0, 0, nullptr, 0, 0,
      nullptr, 0, Whbuf, D, (long long)N * D, N, D);

  k_e1e2<<<(H1 * N + 255) / 256, 256, 0, stream>>>(Whbuf, a1, e1b, e2b, D, N, H1);
  k_attn_topk<<<dim3(N, H1), 32, 0, stream>>>(e1b, e2b, nidx, ncnt, tidx, tval, N);

  hipMemsetAsync(cbuf, 0, (size_t)H1 * N * D * 4, stream);
  hipMemsetAsync(h1cat, 0, (size_t)N * H1 * D * 4, stream);

  for (int t = 0; t < KN; ++t) {
    // z = (Wh[idx_t] * val_t) @ Wi1 + h @ Whh1 + b1   (fused dual-A GEMM)
    k_gemm_wmma<<<dim3(256 / 32, N / 16, H1), 32, 0, stream>>>(
        Whbuf, D, (long long)N * D, Wi1p, D, (long long)256 * D,
        tidx + t, tval + t, (long long)N * KN, KN,
        h1cat, H1 * D, D, Wh1p, D, (long long)256 * D,
        b1, 256, zbuf, 256, (long long)N * 256, N, 256);
    k_lstm_gates<<<(H1 * N * D + 255) / 256, 256, 0, stream>>>(
        zbuf, cbuf, h1cat, N, H1, H1 * D, D);
  }
  k_elu<<<(N * H1 * D + 255) / 256, 256, 0, stream>>>(h1cat, N * H1 * D);

  // ================= Layer 2 (4 heads) =================
  k_gemm_wmma<<<dim3(64 / 32, N / 16, H2), 32, 0, stream>>>(
      h1cat, H1 * D, 0, W2p, H1 * D, (long long)64 * (H1 * D),
      nullptr, nullptr, 0, 0,
      nullptr, 0, 0, nullptr, 0, 0,
      nullptr, 0, Whbuf, D, (long long)N * D, N, D);

  k_e1e2<<<(H2 * N + 255) / 256, 256, 0, stream>>>(Whbuf, a2, e1b, e2b, D, N, H2);
  k_attn_topk<<<dim3(N, H2), 32, 0, stream>>>(e1b, e2b, nidx, ncnt, tidx, tval, N);

  hipMemsetAsync(cbuf, 0, (size_t)H2 * N * D * 4, stream);
  hipMemsetAsync(h2cat, 0, (size_t)N * H2 * D * 4, stream);

  for (int t = 0; t < KN; ++t) {
    k_gemm_wmma<<<dim3(256 / 32, N / 16, H2), 32, 0, stream>>>(
        Whbuf, D, (long long)N * D, Wi2p, D, (long long)256 * D,
        tidx + t, tval + t, (long long)N * KN, KN,
        h2cat, H2 * D, D, Wh2p, D, (long long)256 * D,
        b2, 256, zbuf, 256, (long long)N * 256, N, 256);
    k_lstm_gates<<<(H2 * N * D + 255) / 256, 256, 0, stream>>>(
        zbuf, cbuf, h2cat, N, H2, H2 * D, D);
  }
  k_elu<<<(N * H2 * D + 255) / 256, 256, 0, stream>>>(h2cat, N * H2 * D);

  // ================= Output head =================
  k_gemm_wmma<<<dim3(64 / 32, N / 16, 1), 32, 0, stream>>>(
      h2cat, H2 * D, 0, Wop, H2 * D, 0,
      nullptr, nullptr, 0, 0,
      nullptr, 0, 0, nullptr, 0, 0,
      nullptr, 0, Whp, NC, 0, N, NC);

  k_e1e2<<<(N + 255) / 256, 256, 0, stream>>>(Whp, ao, e1b, e2b, NC, N, 1);
  k_final<<<N, 32, 0, stream>>>(Whp, e1b, e2b, nidx, ncnt, out, N, NC);
}
